// Encoder_59485297050149
// MI455X (gfx1250) — matrix operand order repaired and checked by reading, not compiled, not run on hardware
//
#include <hip/hip_runtime.h>

// ---------------- types ----------------
typedef __bf16 bf16_t;
typedef __attribute__((ext_vector_type(16))) __bf16 v16bf;
typedef __attribute__((ext_vector_type(8)))  __bf16 v8bf;
typedef __attribute__((ext_vector_type(8)))  float  v8f;

#define LDT 40   // padded LDS row (bf16 elems): 80B rows keep 16B alignment for b128 LDS ops

__device__ __forceinline__ v8f wmma_bf16(v16bf a, v16bf b, v8f c) {
  // D = A(16x32 bf16) * B(32x16 bf16) + C(16x16 f32)
  return __builtin_amdgcn_wmma_f32_16x16x32_bf16(false, a, false, b, (short)0, c, false, false);
}

// A-fragment: lane (lidx, lhalf) holds row m=lidx, K = lhalf*8+{0..7} and 16+lhalf*8+{0..7}
__device__ __forceinline__ v16bf frag_a_ld(const bf16_t* tile, int ld, int lidx, int lhalf) {
  const bf16_t* p = tile + lidx * ld + lhalf * 8;
  v8bf lo = *(const v8bf*)(p);
  v8bf hi = *(const v8bf*)(p + 16);
  return __builtin_shufflevector(lo, hi, 0,1,2,3,4,5,6,7,8,9,10,11,12,13,14,15);
}

// B-fragment from [N][K]-major storage: lane n=lidx holds K = lhalf*8+{0..7} and 16+lhalf*8+{0..7}
__device__ __forceinline__ v16bf frag_b_ld(const bf16_t* tile, int ld, int lidx, int lhalf) {
  const bf16_t* p = tile + lidx * ld + lhalf * 8;
  v8bf lo = *(const v8bf*)(p);
  v8bf hi = *(const v8bf*)(p + 16);
  return __builtin_shufflevector(lo, hi, 0,1,2,3,4,5,6,7,8,9,10,11,12,13,14,15);
}

// ---- CDNA5 async copy: global -> LDS (ASYNCcnt-tracked, no VGPR transit) ----
// lds_byte_off = per-lane byte offset within the kernel's static LDS block (base 0).
// lds_escape   = generic pointer to the LDS block: passing it as an asm operand makes
//                the LDS object escape, so the "memory" clobber is known to cover it
//                (otherwise LLVM deletes the dependent ds_load fragment reads).
__device__ __forceinline__ void async_cp_b128(unsigned lds_byte_off, const void* gaddr,
                                              const void* lds_escape) {
  asm volatile("global_load_async_to_lds_b128 %0, %1, off"
               :: "v"(lds_byte_off), "v"(gaddr), "v"(lds_escape)
               : "memory");
}
template <int N>
__device__ __forceinline__ void wait_async() {
#if __has_builtin(__builtin_amdgcn_s_wait_asynccnt)
  __builtin_amdgcn_s_wait_asynccnt(N);
#else
  if constexpr (N == 0)      asm volatile("s_wait_asynccnt 0" ::: "memory");
  else if constexpr (N == 4) asm volatile("s_wait_asynccnt 4" ::: "memory");
  else                       asm volatile("s_wait_asynccnt 5" ::: "memory");
#endif
}

// ---------------- utility kernels ----------------
__global__ void f32_to_bf16_kernel(const float* __restrict__ s, bf16_t* __restrict__ d, size_t n) {
  size_t i = (size_t)blockIdx.x * blockDim.x + threadIdx.x;
  if (i < n) d[i] = (bf16_t)s[i];
}

__global__ __launch_bounds__(256)
void rmsnorm_bf16_kernel(const float* __restrict__ X, const float* __restrict__ g,
                         bf16_t* __restrict__ Y, int D, float eps) {
  const size_t row = blockIdx.x;
  const float* x = X + row * (size_t)D;
  float ss = 0.f;
  for (int i = threadIdx.x; i < D; i += 256) { float v = x[i]; ss += v * v; }
  #pragma unroll
  for (int off = 16; off >= 1; off >>= 1) ss += __shfl_xor(ss, off, 32);
  __shared__ float red[8];
  if ((threadIdx.x & 31) == 0) red[threadIdx.x >> 5] = ss;
  __syncthreads();
  float tot = red[0];
  #pragma unroll
  for (int i = 1; i < 8; ++i) tot += red[i];
  const float inv = rsqrtf(tot / (float)D + eps);
  bf16_t* y = Y + row * (size_t)D;
  for (int i = threadIdx.x; i < D; i += 256) y[i] = (bf16_t)(g[i] * (x[i] * inv));
}

// ---------------- WMMA GEMM: C[M,N] = A[M,K] * B[N,K]^T ----------------
// Block tile 128x128, 8 waves, wave tile 32x64 (8 WMMA / K-step / wave).
// Double-buffered LDS, async global->LDS copies overlap the WMMAs of the
// previous tile (4 async ops/wave/tile; async loads complete in order, so
// s_wait_asynccnt <= 4 guarantees the older tile is resident).
// MODE 0: store bf16 C.  MODE 1: store f32 C = residual + acc.
template <int MODE>
__global__ __launch_bounds__(256)
void gemm_bf16_kernel(const bf16_t* __restrict__ A, const bf16_t* __restrict__ Bm,
                      int M, int N, int K,
                      bf16_t* __restrict__ Cb, float* __restrict__ Cf,
                      const float* __restrict__ R) {
  constexpr int TILE = 2 * 128 * LDT;              // elems per buffer: As | Bs
  __shared__ bf16_t smem[2 * TILE];

  const int tid = threadIdx.x;
  const int lane = tid & 31, wid = tid >> 5;
  const int lidx = lane & 15, lhalf = lane >> 4;
  const int wr = wid >> 1, wc = wid & 1;           // 4x2 wave grid -> 128x128 block tile
  const int m_blk = blockIdx.y * 128;
  const int n_blk = blockIdx.x * 128;

  // staging: both tiles are 128 rows x 32 cols; 16 elems (2 x b128) per thread each
  const int srow = tid >> 1, scol = (tid & 1) * 16;
  const unsigned aoff = (unsigned)((srow * LDT + scol) * 2);               // bytes
  const unsigned boff = (unsigned)(((128 * LDT) + srow * LDT + scol) * 2);
  const bf16_t* gA = A  + (size_t)(m_blk + srow) * K + scol;
  const bf16_t* gB = Bm + (size_t)(n_blk + srow) * K + scol;

  auto issue = [&](int k0, int p) {
    const unsigned pb = (unsigned)(p * TILE * 2);
    async_cp_b128(pb + aoff,      gA + k0,     smem);
    async_cp_b128(pb + aoff + 16, gA + k0 + 8, smem);
    async_cp_b128(pb + boff,      gB + k0,     smem);
    async_cp_b128(pb + boff + 16, gB + k0 + 8, smem);
  };

  v8f acc[2][4] = {};

  issue(0, 0);
  int p = 0;
  for (int k0 = 0; k0 < K; k0 += 32) {
    if (k0 + 32 < K) { issue(k0 + 32, p ^ 1); wait_async<4>(); }
    else             { wait_async<0>(); }
    __syncthreads();                               // tile p resident for all waves

    const bf16_t* As = smem + p * TILE;
    const bf16_t* Bs = As + 128 * LDT;
    v16bf af0 = frag_a_ld(As + (wr * 32 +  0) * LDT, LDT, lidx, lhalf);
    v16bf af1 = frag_a_ld(As + (wr * 32 + 16) * LDT, LDT, lidx, lhalf);
    v16bf bfv[4];
    #pragma unroll
    for (int ni = 0; ni < 4; ++ni)
      bfv[ni] = frag_b_ld(Bs + (wc * 64 + ni * 16) * LDT, LDT, lidx, lhalf);
    #pragma unroll
    for (int ni = 0; ni < 4; ++ni) acc[0][ni] = wmma_bf16(af0, bfv[ni], acc[0][ni]);
    #pragma unroll
    for (int ni = 0; ni < 4; ++ni) acc[1][ni] = wmma_bf16(af1, bfv[ni], acc[1][ni]);

    __syncthreads();                               // all reads of buffer p done
    p ^= 1;
  }

  #pragma unroll
  for (int mi = 0; mi < 2; ++mi)
    #pragma unroll
    for (int ni = 0; ni < 4; ++ni) {
      const int mbase = m_blk + wr * 32 + mi * 16;
      const int nbase = n_blk + wc * 64 + ni * 16;
      #pragma unroll
      for (int r = 0; r < 8; ++r) {
        const size_t idx = (size_t)(mbase + r + 8 * lhalf) * N + nbase + lidx;
        const float v = acc[mi][ni][r];
        if constexpr (MODE == 0) Cb[idx] = (bf16_t)v;
        else                     Cf[idx] = R[idx] + v;
      }
    }
}

// ---------------- dual GEMM + SwiGLU: U = silu(A*W1^T) * (A*W3^T), bf16 ----------------
// Block tile 64x128, 8 waves, wave tile 32x32 per matrix (two accumulator sets).
// Same double-buffered async pipeline (5 async ops/wave/tile).
__global__ __launch_bounds__(256)
void dualgemm_swiglu_kernel(const bf16_t* __restrict__ A,
                            const bf16_t* __restrict__ B1, const bf16_t* __restrict__ B3,
                            int M, int N, int K, bf16_t* __restrict__ U) {
  constexpr int TILE = (64 + 128 + 128) * LDT;     // elems per buffer: As | Bs1 | Bs3
  __shared__ bf16_t smem[2 * TILE];

  const int tid = threadIdx.x;
  const int lane = tid & 31, wid = tid >> 5;
  const int lidx = lane & 15, lhalf = lane >> 4;
  const int wr = wid >> 2, wc = wid & 3;           // 2x4 wave grid -> 64x128 block tile
  const int m_blk = blockIdx.y * 64;
  const int n_blk = blockIdx.x * 128;

  const int arow = tid >> 2, acol = (tid & 3) * 8;  // A tile 64x32: 1 x b128 per thread
  const int brow = tid >> 1, bcol = (tid & 1) * 16; // B tiles 128x32: 2 x b128 per thread
  const unsigned aoff  = (unsigned)((arow * LDT + acol) * 2);
  const unsigned b1off = (unsigned)(((64 * LDT) + brow * LDT + bcol) * 2);
  const unsigned b3off = (unsigned)((((64 + 128) * LDT) + brow * LDT + bcol) * 2);
  const bf16_t* gA  = A  + (size_t)(m_blk + arow) * K + acol;
  const bf16_t* gB1 = B1 + (size_t)(n_blk + brow) * K + bcol;
  const bf16_t* gB3 = B3 + (size_t)(n_blk + brow) * K + bcol;

  auto issue = [&](int k0, int p) {
    const unsigned pb = (unsigned)(p * TILE * 2);
    async_cp_b128(pb + aoff,       gA  + k0,     smem);
    async_cp_b128(pb + b1off,      gB1 + k0,     smem);
    async_cp_b128(pb + b1off + 16, gB1 + k0 + 8, smem);
    async_cp_b128(pb + b3off,      gB3 + k0,     smem);
    async_cp_b128(pb + b3off + 16, gB3 + k0 + 8, smem);
  };

  v8f acc1[2][2] = {}, acc3[2][2] = {};

  issue(0, 0);
  int p = 0;
  for (int k0 = 0; k0 < K; k0 += 32) {
    if (k0 + 32 < K) { issue(k0 + 32, p ^ 1); wait_async<5>(); }
    else             { wait_async<0>(); }
    __syncthreads();

    const bf16_t* As  = smem + p * TILE;
    const bf16_t* Bs1 = As + 64 * LDT;
    const bf16_t* Bs3 = As + (64 + 128) * LDT;
    #pragma unroll
    for (int mi = 0; mi < 2; ++mi) {
      v16bf af = frag_a_ld(As + (wr * 32 + mi * 16) * LDT, LDT, lidx, lhalf);
      #pragma unroll
      for (int ni = 0; ni < 2; ++ni) {
        v16bf bf1 = frag_b_ld(Bs1 + (wc * 32 + ni * 16) * LDT, LDT, lidx, lhalf);
        v16bf bf3 = frag_b_ld(Bs3 + (wc * 32 + ni * 16) * LDT, LDT, lidx, lhalf);
        acc1[mi][ni] = wmma_bf16(af, bf1, acc1[mi][ni]);
        acc3[mi][ni] = wmma_bf16(af, bf3, acc3[mi][ni]);
      }
    }

    __syncthreads();
    p ^= 1;
  }

  #pragma unroll
  for (int mi = 0; mi < 2; ++mi)
    #pragma unroll
    for (int ni = 0; ni < 2; ++ni) {
      const int mbase = m_blk + wr * 32 + mi * 16;
      const int nbase = n_blk + wc * 32 + ni * 16;
      #pragma unroll
      for (int r = 0; r < 8; ++r) {
        const float a = acc1[mi][ni][r];
        const float b = acc3[mi][ni][r];
        const float s = a / (1.0f + __expf(-a));   // silu
        U[(size_t)(mbase + r + 8 * lhalf) * N + nbase + lidx] = (bf16_t)(s * b);
      }
    }
}

// ---------------- flash attention (bf16 QKV in [B,S,H*Dk] layout) ----------------
// grid: (S/64, B*H); block: 128 threads (4 waves), each wave owns 16 query rows.
__global__ __launch_bounds__(128)
void flash_attn_kernel(const bf16_t* __restrict__ Q, const bf16_t* __restrict__ Kg,
                       const bf16_t* __restrict__ Vg, const int* __restrict__ mask,
                       bf16_t* __restrict__ O, int Bsz, int S, int H) {
  const int Dk = 64;
  const int bh = blockIdx.y;
  const int b  = bh / H, h = bh % H;
  const int D  = H * Dk;
  const int tid = threadIdx.x, lane = tid & 31, wid = tid >> 5;
  const int lidx = lane & 15, lhalf = lane >> 4;
  const int qrow0 = blockIdx.x * 64 + wid * 16;

  __shared__ bf16_t Vs[64 * LDT];        // V transposed: [dk][key]
  __shared__ bf16_t Ps[4][16 * LDT];     // per-wave P staging: [m][key]

  // persistent Q fragments (A-layout loads straight from global)
  const bf16_t* qbase = Q + ((size_t)(b * S + qrow0)) * D + h * Dk;
  v16bf aq0 = frag_a_ld(qbase,      D, lidx, lhalf);   // K 0..31
  v16bf aq1 = frag_a_ld(qbase + 32, D, lidx, lhalf);   // K 32..63

  v8f o[4] = {};
  float m8[8], l8[8];
  #pragma unroll
  for (int r = 0; r < 8; ++r) { m8[r] = -1e30f; l8[r] = 0.f; }

  const int vkey = tid >> 2, vdkb = (tid & 3) * 16;    // V staging indices

  for (int kv0 = 0; kv0 < S; kv0 += 32) {
    // stage V^T cooperatively
    {
      const bf16_t* vp = Vg + ((size_t)(b * S + kv0 + vkey)) * D + h * Dk + vdkb;
      bf16_t tmp[16];
      #pragma unroll
      for (int j = 0; j < 16; ++j) tmp[j] = vp[j];
      __syncthreads();                                  // prev iter done reading Vs
      #pragma unroll
      for (int j = 0; j < 16; ++j) Vs[(vdkb + j) * LDT + vkey] = tmp[j];
      __syncthreads();
    }

    // scores: S16x32 = Q(16x64) * K^T  (B-frags straight from global K rows)
    v8f sc[2];
    #pragma unroll
    for (int ni = 0; ni < 2; ++ni) {
      const bf16_t* kbase = Kg + ((size_t)(b * S + kv0 + ni * 16)) * D + h * Dk;
      v8f a = {};
      a = wmma_bf16(aq0, frag_b_ld(kbase,      D, lidx, lhalf), a);
      a = wmma_bf16(aq1, frag_b_ld(kbase + 32, D, lidx, lhalf), a);
      sc[ni] = a;
    }

    // scale + mask
    #pragma unroll
    for (int ni = 0; ni < 2; ++ni)
      #pragma unroll
      for (int r = 0; r < 8; ++r) {
        const int qr = qrow0 + r + 8 * lhalf;
        const int kc = kv0 + ni * 16 + lidx;
        const float s = sc[ni][r] * 0.125f;             // 1/sqrt(64)
        const int mv = mask[((size_t)b * S + qr) * S + kc];
        sc[ni][r] = (mv == 0) ? -1e9f : s;
      }

    // online softmax (row = 16 lanes of one half-wave)
    #pragma unroll
    for (int r = 0; r < 8; ++r) {
      float mx = fmaxf(sc[0][r], sc[1][r]);
      #pragma unroll
      for (int off = 8; off >= 1; off >>= 1) mx = fmaxf(mx, __shfl_xor(mx, off, 32));
      const float mnew  = fmaxf(m8[r], mx);
      const float alpha = __expf(m8[r] - mnew);
      const float p0 = __expf(sc[0][r] - mnew);
      const float p1 = __expf(sc[1][r] - mnew);
      sc[0][r] = p0; sc[1][r] = p1;
      float rs = p0 + p1;
      #pragma unroll
      for (int off = 8; off >= 1; off >>= 1) rs += __shfl_xor(rs, off, 32);
      l8[r] = l8[r] * alpha + rs;
      m8[r] = mnew;
      #pragma unroll
      for (int ni = 0; ni < 4; ++ni) o[ni][r] *= alpha;
    }

    // restage P (C-layout -> A-layout) through per-wave LDS
    bf16_t* pw = &Ps[wid][0];
    #pragma unroll
    for (int ni = 0; ni < 2; ++ni)
      #pragma unroll
      for (int r = 0; r < 8; ++r)
        pw[(r + 8 * lhalf) * LDT + ni * 16 + lidx] = (bf16_t)sc[ni][r];

    // O += P(16x32) * V(32x64)
    v16bf ap = frag_a_ld(pw, LDT, lidx, lhalf);
    #pragma unroll
    for (int ni = 0; ni < 4; ++ni) {
      v16bf bv = frag_b_ld(Vs + (ni * 16) * LDT, LDT, lidx, lhalf);
      o[ni] = wmma_bf16(ap, bv, o[ni]);
    }
  }

  // normalize + store bf16
  #pragma unroll
  for (int ni = 0; ni < 4; ++ni)
    #pragma unroll
    for (int r = 0; r < 8; ++r) {
      const int qr = qrow0 + r + 8 * lhalf;
      const float ov = o[ni][r] / l8[r];
      O[((size_t)(b * S + qr)) * D + h * Dk + ni * 16 + lidx] = (bf16_t)ov;
    }
}

// ---------------- launch ----------------
extern "C" void kernel_launch(void* const* d_in, const int* in_sizes, int n_in,
                              void* d_out, int out_size, void* d_ws, size_t ws_size,
                              hipStream_t stream) {
  (void)in_sizes; (void)n_in; (void)out_size; (void)ws_size;
  const int Bsz = 2, S = 2048, D = 768, H = 12, F = 3072;
  const int M = Bsz * S;

  const float* x      = (const float*)d_in[0];
  const int*   mask   = (const int*)  d_in[1];
  const float* wq     = (const float*)d_in[2];
  const float* wk     = (const float*)d_in[3];
  const float* wv     = (const float*)d_in[4];
  const float* wo     = (const float*)d_in[5];
  const float* w1     = (const float*)d_in[6];
  const float* w2     = (const float*)d_in[7];
  const float* w3     = (const float*)d_in[8];
  const float* g_attn = (const float*)d_in[9];
  const float* g_ffn  = (const float*)d_in[10];
  float* out = (float*)d_out;

  // bump allocator over workspace (256B aligned)
  char* ws = (char*)d_ws;
  auto alloc = [&](size_t bytes) -> char* {
    char* p = ws;
    ws += (bytes + 255) & ~(size_t)255;
    return p;
  };
  bf16_t* wqb = (bf16_t*)alloc((size_t)D * D * 2);
  bf16_t* wkb = (bf16_t*)alloc((size_t)D * D * 2);
  bf16_t* wvb = (bf16_t*)alloc((size_t)D * D * 2);
  bf16_t* wob = (bf16_t*)alloc((size_t)D * D * 2);
  bf16_t* w1b = (bf16_t*)alloc((size_t)F * D * 2);
  bf16_t* w3b = (bf16_t*)alloc((size_t)F * D * 2);
  bf16_t* w2b = (bf16_t*)alloc((size_t)D * F * 2);
  bf16_t* xn  = (bf16_t*)alloc((size_t)M * D * 2);
  bf16_t* qb  = (bf16_t*)alloc((size_t)M * D * 2);
  bf16_t* kb  = (bf16_t*)alloc((size_t)M * D * 2);
  bf16_t* vb  = (bf16_t*)alloc((size_t)M * D * 2);
  bf16_t* ab  = (bf16_t*)alloc((size_t)M * D * 2);   // attention output (bf16)
  float*  hb  = (float*) alloc((size_t)M * D * 4);   // h = x + attn@Wo^T (f32)
  bf16_t* hn  = (bf16_t*)alloc((size_t)M * D * 2);
  bf16_t* ub  = (bf16_t*)alloc((size_t)M * F * 2);

  auto cvt = [&](const float* s, bf16_t* d, size_t n) {
    f32_to_bf16_kernel<<<(unsigned)((n + 255) / 256), 256, 0, stream>>>(s, d, n);
  };
  cvt(wq, wqb, (size_t)D * D);
  cvt(wk, wkb, (size_t)D * D);
  cvt(wv, wvb, (size_t)D * D);
  cvt(wo, wob, (size_t)D * D);
  cvt(w1, w1b, (size_t)F * D);
  cvt(w3, w3b, (size_t)F * D);
  cvt(w2, w2b, (size_t)D * F);

  // 1) pre-norm
  rmsnorm_bf16_kernel<<<M, 256, 0, stream>>>(x, g_attn, xn, D, 1e-5f);

  // 2) QKV projections (block tile 128x128)
  dim3 gDD(D / 128, M / 128);
  gemm_bf16_kernel<0><<<gDD, 256, 0, stream>>>(xn, wqb, M, D, D, qb, nullptr, nullptr);
  gemm_bf16_kernel<0><<<gDD, 256, 0, stream>>>(xn, wkb, M, D, D, kb, nullptr, nullptr);
  gemm_bf16_kernel<0><<<gDD, 256, 0, stream>>>(xn, wvb, M, D, D, vb, nullptr, nullptr);

  // 3) attention (flash, online softmax)
  dim3 gFA(S / 64, Bsz * H);
  flash_attn_kernel<<<gFA, 128, 0, stream>>>(qb, kb, vb, mask, ab, Bsz, S, H);

  // 4) output projection + residual -> h (f32)
  gemm_bf16_kernel<1><<<gDD, 256, 0, stream>>>(ab, wob, M, D, D, nullptr, hb, x);

  // 5) FFN pre-norm
  rmsnorm_bf16_kernel<<<M, 256, 0, stream>>>(hb, g_ffn, hn, D, 1e-5f);

  // 6) fused gate/up GEMM + SwiGLU (block tile 64x128)
  dim3 gDF(F / 128, M / 64);
  dualgemm_swiglu_kernel<<<gDF, 256, 0, stream>>>(hn, w1b, w3b, M, F, D, ub);

  // 7) down projection + residual -> out (f32)
  gemm_bf16_kernel<1><<<gDD, 256, 0, stream>>>(ub, w2b, M, D, F, nullptr, out, hb);
}
